// Model_37838661878530
// MI455X (gfx1250) — matrix operand order repaired and checked by reading, not compiled, not run on hardware
//
#include <hip/hip_runtime.h>
#include <hip/hip_bf16.h>
#include <math.h>

// Model dims
#define D_   4096
#define H_   32
#define HD_  128
#define FF_  11008
#define L_   2
#define S_   64
#define B_   32
#define MT_  (B_ * S_)          // 2048 tokens
#define EPS_ 1e-5f

typedef __attribute__((ext_vector_type(16))) __bf16 v16bf;
typedef __attribute__((ext_vector_type(8)))  float  v8f;
typedef unsigned int u32x4 __attribute__((ext_vector_type(4)));
typedef int          i32x4 __attribute__((ext_vector_type(4)));
typedef int          i32x8 __attribute__((ext_vector_type(8)));

// ---------------------------------------------------------------------------
// TDM 2D tile load: global (row-major, elem stride_d0 per row) -> LDS.
// Descriptor packing per CDNA5 ISA ch. 8 (group0 128b, group1 256b).
// data_size = 4 bytes. 2D: groups 2/3 zero.
// ---------------------------------------------------------------------------
__device__ inline void tdm_load_2d(unsigned lds_off, const float* gptr,
                                   unsigned tensor_d0, unsigned tensor_d1,
                                   unsigned stride_d0, unsigned tile_d0,
                                   unsigned tile_d1) {
  unsigned long long ga = (unsigned long long)(uintptr_t)gptr;
  u32x4 g0;
  g0[0] = 1u;                                            // count=1 (valid), user mode
  g0[1] = lds_off;                                       // lds_addr (bytes)
  g0[2] = (unsigned)(ga & 0xFFFFFFFFu);                  // global_addr[31:0]
  g0[3] = (unsigned)((ga >> 32) & 0x01FFFFFFu) | (2u << 30); // addr[56:32] | type=2
  i32x8 g1;
  g1[0] = (int)(2u << 16);                               // wg_mask=0, data_size=4B
  g1[1] = (int)((tensor_d0 & 0xFFFFu) << 16);            // tensor_dim0[15:0] @ bit48
  g1[2] = (int)((tensor_d0 >> 16) | ((tensor_d1 & 0xFFFFu) << 16)); // d0 hi | d1 lo
  g1[3] = (int)((tensor_d1 >> 16) | (tile_d0 << 16));    // d1 hi | tile_dim0
  g1[4] = (int)(tile_d1 & 0xFFFFu);                      // tile_dim1 | tile_dim2=0
  g1[5] = (int)stride_d0;                                // tensor_dim0_stride[31:0]
  g1[6] = 0;                                             // stride hi16 | dim1_stride lo (2D)
  g1[7] = 0;
  i32x4 z4 = {0, 0, 0, 0};
#if defined(__clang_major__) && (__clang_major__ >= 23)
  i32x8 z8 = {0, 0, 0, 0, 0, 0, 0, 0};
  __builtin_amdgcn_tensor_load_to_lds(g0, g1, z4, z4, z8, 0);
#else
  __builtin_amdgcn_tensor_load_to_lds(g0, g1, z4, z4, 0);
#endif
}

// ---------------------------------------------------------------------------
// WMMA fragment loaders from LDS fp32 tiles -> bf16 fragments.
// A 16x32 bf16 layout: lanes 0-15 row M=lane, K = 0..15 contiguous;
//                      lanes 16-31 row M=lane-16, K = 16..31.
// B 32x16 bf16 layout: lanes 0-15 col N=lane, K = 0..15 contiguous;
//                      lanes 16-31 col N=lane-16, K = 16..31.
// ---------------------------------------------------------------------------
__device__ inline v16bf frag_a_lds(const float* rowptr, int kb) {
  const float4* p = (const float4*)(rowptr + kb);
  float4 x0 = p[0], x1 = p[1];
  const float4* q = (const float4*)(rowptr + kb + 8);
  float4 y0 = q[0], y1 = q[1];
  v16bf r;
  r[0]=(__bf16)x0.x;  r[1]=(__bf16)x0.y;  r[2]=(__bf16)x0.z;  r[3]=(__bf16)x0.w;
  r[4]=(__bf16)x1.x;  r[5]=(__bf16)x1.y;  r[6]=(__bf16)x1.z;  r[7]=(__bf16)x1.w;
  r[8]=(__bf16)y0.x;  r[9]=(__bf16)y0.y;  r[10]=(__bf16)y0.z; r[11]=(__bf16)y0.w;
  r[12]=(__bf16)y1.x; r[13]=(__bf16)y1.y; r[14]=(__bf16)y1.z; r[15]=(__bf16)y1.w;
  return r;
}

__device__ inline v16bf frag_b_lds(const float* rowptr, int kb16) {
  const float4* p = (const float4*)(rowptr + kb16);
  float4 x0 = p[0], x1 = p[1], x2 = p[2], x3 = p[3];
  v16bf r;
  r[0]=(__bf16)x0.x;  r[1]=(__bf16)x0.y;  r[2]=(__bf16)x0.z;  r[3]=(__bf16)x0.w;
  r[4]=(__bf16)x1.x;  r[5]=(__bf16)x1.y;  r[6]=(__bf16)x1.z;  r[7]=(__bf16)x1.w;
  r[8]=(__bf16)x2.x;  r[9]=(__bf16)x2.y;  r[10]=(__bf16)x2.z; r[11]=(__bf16)x2.w;
  r[12]=(__bf16)x3.x; r[13]=(__bf16)x3.y; r[14]=(__bf16)x3.z; r[15]=(__bf16)x3.w;
  return r;
}

// ---------------------------------------------------------------------------
// GEMM: C[M,N] = A[M,K] @ W[N,K]^T (+ optional residual). fp32 in/out,
// bf16 WMMA (fp32 accumulate). Block = 256 threads (8 waves).
// Workgroup tile 128(M) x 64(N); wave w owns rows [w*16, +16).
// K stepped by 32 with double-buffered LDS tiles fed by the TDM:
//   A tile 128x32 fp32 (16KB), W tile 64x32 fp32 (8KB), x2 buffers = 48KB.
// Wave 0 issues tensor_load_to_lds for the next K-slab, all waves compute
// from the current slab (ds_load_b128 + cvt -> 4x v_wmma per wave), then
// wave 0 waits TENSORcnt==0 and the block barriers.
// M%128==0, N%64==0, K%32==0 for all calls here.
// ---------------------------------------------------------------------------
__global__ __launch_bounds__(256)
void gemm_bf16_tn(const float* __restrict__ A, const float* __restrict__ W,
                  const float* __restrict__ res, float* __restrict__ C,
                  int M, int N, int K) {
  __shared__ float As[2][128 * 32];
  __shared__ float Ws[2][64 * 32];

  const int lane = threadIdx.x & 31;
  const int wave = threadIdx.x >> 5;
  const int m0   = blockIdx.y * 128 + wave * 16;
  const int n0   = blockIdx.x * 64;
  const int lo   = lane & 15;
  const int hi8  = (lane >> 4) << 3;   // unused; A fragment K sub-offset is hi16 too
  const int hi16 = (lane >> 4) << 4;   // B fragment K sub-offset (0 or 16)

  const float* Abase = A + (size_t)(blockIdx.y * 128) * K;
  const float* Wbase = W + (size_t)n0 * K;
  const unsigned a_lds[2] = { (unsigned)(uintptr_t)&As[0][0],
                              (unsigned)(uintptr_t)&As[1][0] };
  const unsigned w_lds[2] = { (unsigned)(uintptr_t)&Ws[0][0],
                              (unsigned)(uintptr_t)&Ws[1][0] };

  v8f acc0 = {}, acc1 = {}, acc2 = {}, acc3 = {};
  const int nk = K >> 5;

  // Prologue: stage K-slab 0 into buffer 0.
  if (wave == 0) {
    tdm_load_2d(a_lds[0], Abase, (unsigned)K, 128u, (unsigned)K, 32u, 128u);
    tdm_load_2d(w_lds[0], Wbase, (unsigned)K, 64u,  (unsigned)K, 32u, 64u);
    __builtin_amdgcn_s_wait_tensorcnt(0);
  }
  __syncthreads();

  for (int kt = 0; kt < nk; ++kt) {
    const int cur = kt & 1;
    const int nxt = cur ^ 1;

    // Async-stage the next K-slab while computing on the current one.
    if (wave == 0 && (kt + 1) < nk) {
      const int k1 = (kt + 1) << 5;
      tdm_load_2d(a_lds[nxt], Abase + k1, (unsigned)K, 128u, (unsigned)K, 32u, 128u);
      tdm_load_2d(w_lds[nxt], Wbase + k1, (unsigned)K, 64u,  (unsigned)K, 32u, 64u);
    }

    const float* as = &As[cur][0] + (size_t)(wave * 16 + lo) * 32;
    const float* ws0 = &Ws[cur][0] + (size_t)(0  + lo) * 32;
    const float* ws1 = &Ws[cur][0] + (size_t)(16 + lo) * 32;
    const float* ws2 = &Ws[cur][0] + (size_t)(32 + lo) * 32;
    const float* ws3 = &Ws[cur][0] + (size_t)(48 + lo) * 32;

    v16bf a  = frag_a_lds(as, hi16);
    v16bf b0 = frag_b_lds(ws0, hi16);
    v16bf b1 = frag_b_lds(ws1, hi16);
    v16bf b2 = frag_b_lds(ws2, hi16);
    v16bf b3 = frag_b_lds(ws3, hi16);
    acc0 = __builtin_amdgcn_wmma_f32_16x16x32_bf16(false, a, false, b0, (short)0, acc0, false, false);
    acc1 = __builtin_amdgcn_wmma_f32_16x16x32_bf16(false, a, false, b1, (short)0, acc1, false, false);
    acc2 = __builtin_amdgcn_wmma_f32_16x16x32_bf16(false, a, false, b2, (short)0, acc2, false, false);
    acc3 = __builtin_amdgcn_wmma_f32_16x16x32_bf16(false, a, false, b3, (short)0, acc3, false, false);

    if (wave == 0) __builtin_amdgcn_s_wait_tensorcnt(0);
    __syncthreads();
  }

  // C/D layout: VGPR r, lanes 0-15 -> M = r, N = lane; lanes 16-31 -> M = 8+r.
  const int rbase = m0 + ((lane >> 4) << 3);
  #pragma unroll
  for (int r = 0; r < 8; ++r) {
    size_t rowoff = (size_t)(rbase + r) * N;
    float v0 = acc0[r], v1 = acc1[r], v2 = acc2[r], v3 = acc3[r];
    if (res) {
      v0 += res[rowoff + n0 +  0 + lo];
      v1 += res[rowoff + n0 + 16 + lo];
      v2 += res[rowoff + n0 + 32 + lo];
      v3 += res[rowoff + n0 + 48 + lo];
    }
    C[rowoff + n0 +  0 + lo] = v0;
    C[rowoff + n0 + 16 + lo] = v1;
    C[rowoff + n0 + 32 + lo] = v2;
    C[rowoff + n0 + 48 + lo] = v3;
  }
}

// ---------------------------------------------------------------------------
// Block reduction helper (256 threads)
// ---------------------------------------------------------------------------
__device__ inline float block_reduce_sum_256(float v, float* red) {
  for (int off = 16; off; off >>= 1) v += __shfl_xor(v, off, 32);
  if ((threadIdx.x & 31) == 0) red[threadIdx.x >> 5] = v;
  __syncthreads();
  if (threadIdx.x == 0) {
    float t = 0.f;
    #pragma unroll
    for (int i = 0; i < 8; ++i) t += red[i];
    red[0] = t;
  }
  __syncthreads();
  return red[0];
}

// ---------------------------------------------------------------------------
// Conv front-end
// ---------------------------------------------------------------------------
__global__ void conv1_pool(const float* __restrict__ x, const float* __restrict__ w,
                           const float* __restrict__ bias, float* __restrict__ out) {
  int i = blockIdx.x * blockDim.x + threadIdx.x;
  if (i >= 32 * 32 * 31 * 31) return;
  int px = i % 31, py = (i / 31) % 31, oc = (i / 961) % 32, b = i / (961 * 32);
  const float* xi = x + (size_t)b * 64 * 64;
  const float* wk = w + oc * 9;
  float bsv = bias[oc];
  float acc = 0.f;
  for (int dy = 0; dy < 2; ++dy)
    for (int dx = 0; dx < 2; ++dx) {
      int cy = 2 * py + dy, cx = 2 * px + dx;
      float s = bsv;
      #pragma unroll
      for (int ky = 0; ky < 3; ++ky)
        #pragma unroll
        for (int kx = 0; kx < 3; ++kx)
          s += wk[ky * 3 + kx] * xi[(cy + ky) * 64 + cx + kx];
      acc += s;
    }
  out[i] = acc * 0.25f;
}

__global__ void conv2_pool(const float* __restrict__ h1, const float* __restrict__ w,
                           const float* __restrict__ bias, float* __restrict__ out) {
  int i = blockIdx.x * blockDim.x + threadIdx.x;
  if (i >= 32 * 64 * 14 * 14) return;
  int px = i % 14, py = (i / 14) % 14, oc = (i / 196) % 64, b = i / (196 * 64);
  float bsv = bias[oc];
  float acc = 0.f;
  for (int dy = 0; dy < 2; ++dy)
    for (int dx = 0; dx < 2; ++dx) {
      int cy = 2 * py + dy, cx = 2 * px + dx;
      float s = bsv;
      for (int ic = 0; ic < 32; ++ic) {
        const float* xi = h1 + ((size_t)b * 32 + ic) * 961;   // 31*31
        const float* wk = w + ((size_t)oc * 32 + ic) * 9;
        #pragma unroll
        for (int ky = 0; ky < 3; ++ky)
          #pragma unroll
          for (int kx = 0; kx < 3; ++kx)
            s += wk[ky * 3 + kx] * xi[(cy + ky) * 31 + cx + kx];
      }
      acc += s;
    }
  out[i] = acc * 0.25f;
}

// tok[b,s] = h2_flat[b,:] . it_w[s,:] + it_b[s]
__global__ __launch_bounds__(256)
void it_kernel(const float* __restrict__ h2, const float* __restrict__ itw,
               const float* __restrict__ itb, float* __restrict__ tok) {
  __shared__ float red[8];
  int b = blockIdx.x >> 6, s = blockIdx.x & 63;
  const float* hr = h2 + (size_t)b * 12544;
  const float* wr = itw + (size_t)s * 12544;
  float ss = 0.f;
  for (int i = threadIdx.x; i < 12544; i += 256) ss += hr[i] * wr[i];
  float tot = block_reduce_sum_256(ss, red);
  if (threadIdx.x == 0) tok[blockIdx.x] = tot + itb[s];
}

// hs[t, d] = tok[t] * token_proj[d]
__global__ void embed_kernel(const float* __restrict__ tok, const float* __restrict__ proj,
                             float* __restrict__ hs) {
  int i = blockIdx.x * blockDim.x + threadIdx.x;
  if (i >= MT_ * D_) return;
  hs[i] = tok[i >> 12] * proj[i & (D_ - 1)];
}

// ---------------------------------------------------------------------------
// RMSNorm: one block (256) per token row
// ---------------------------------------------------------------------------
__global__ __launch_bounds__(256)
void rmsnorm_kernel(const float* __restrict__ x, const float* __restrict__ w,
                    float* __restrict__ out) {
  __shared__ float red[8];
  const float* xr = x + (size_t)blockIdx.x * D_;
  float ss = 0.f;
  for (int i = threadIdx.x; i < D_; i += 256) { float t = xr[i]; ss += t * t; }
  float tot = block_reduce_sum_256(ss, red);
  float rms = rsqrtf(tot / (float)D_ + EPS_);
  float* orow = out + (size_t)blockIdx.x * D_;
  for (int i = threadIdx.x; i < D_; i += 256) orow[i] = xr[i] * rms * w[i];
}

// ---------------------------------------------------------------------------
// RoPE tables + in-place application to q and k
// ---------------------------------------------------------------------------
__global__ void rope_tables(float* __restrict__ cost, float* __restrict__ sint) {
  int i = blockIdx.x * blockDim.x + threadIdx.x;   // s*64 + j
  if (i >= S_ * 64) return;
  int s = i >> 6, j = i & 63;
  float inv = powf(10000.0f, -(float)(2 * j) / (float)HD_);
  float f = (float)s * inv;
  float c = cosf(f), sn = sinf(f);
  cost[s * HD_ + j] = c;  cost[s * HD_ + 64 + j] = c;
  sint[s * HD_ + j] = sn; sint[s * HD_ + 64 + j] = sn;
}

__global__ void rope_apply(float* __restrict__ q, float* __restrict__ k,
                           const float* __restrict__ cost, const float* __restrict__ sint) {
  int i = blockIdx.x * blockDim.x + threadIdx.x;   // t*2048 + h*64 + j
  if (i >= MT_ * H_ * 64) return;
  int j = i & 63, h = (i >> 6) & (H_ - 1), t = i >> 11;
  int s = t & (S_ - 1);
  size_t base = (size_t)t * D_ + (size_t)h * HD_;
  float c = cost[s * HD_ + j], sn = sint[s * HD_ + j];
  float q1 = q[base + j], q2 = q[base + 64 + j];
  q[base + j]      = q1 * c - q2 * sn;
  q[base + 64 + j] = q2 * c + q1 * sn;
  float k1 = k[base + j], k2 = k[base + 64 + j];
  k[base + j]      = k1 * c - k2 * sn;
  k[base + 64 + j] = k2 * c + k1 * sn;
}

// ---------------------------------------------------------------------------
// Attention: one wave per (b, h, qi). Lanes own 2 keys each for scores,
// then parallelize over head-dim for attn @ v. S = 64, HD = 128.
// ---------------------------------------------------------------------------
__global__ __launch_bounds__(32)
void attn_kernel(const float* __restrict__ q, const float* __restrict__ k,
                 const float* __restrict__ v, float* __restrict__ o) {
  const int idx  = blockIdx.x;                 // ((b*H)+h)*S + qi
  const int qi   = idx & (S_ - 1);
  const int h    = (idx >> 6) & (H_ - 1);
  const int b    = idx >> 11;
  const int lane = threadIdx.x;
  const float scale = 0.08838834764831845f;    // 1/sqrt(128)

  const float* qrow = q + ((size_t)(b * S_ + qi) * D_ + (size_t)h * HD_);
  const float* k0r  = k + ((size_t)(b * S_ + lane) * D_ + (size_t)h * HD_);
  const float* k1r  = k0r + (size_t)32 * D_;

  float s0 = 0.f, s1 = 0.f;
  for (int d = 0; d < HD_; ++d) {
    float qd = qrow[d];                        // uniform across lanes
    s0 += qd * k0r[d];
    s1 += qd * k1r[d];
  }
  s0 *= scale; s1 *= scale;
  if (lane > qi)      s0 = -1e30f;
  if (lane + 32 > qi) s1 = -1e30f;

  float m = fmaxf(s0, s1);
  for (int off = 16; off; off >>= 1) m = fmaxf(m, __shfl_xor(m, off, 32));
  float p0 = expf(s0 - m), p1 = expf(s1 - m);
  float sum = p0 + p1;
  for (int off = 16; off; off >>= 1) sum += __shfl_xor(sum, off, 32);
  float invs = 1.f / sum;
  p0 *= invs; p1 *= invs;

  float o0 = 0.f, o1 = 0.f, o2 = 0.f, o3 = 0.f;
  const float* vb = v + ((size_t)(b * S_) * D_ + (size_t)h * HD_);
  for (int kj = 0; kj < 32; ++kj) {
    float p = __shfl(p0, kj, 32);
    const float* vr = vb + (size_t)kj * D_;
    o0 += p * vr[lane]; o1 += p * vr[32 + lane];
    o2 += p * vr[64 + lane]; o3 += p * vr[96 + lane];
  }
  for (int kj = 0; kj < 32; ++kj) {
    float p = __shfl(p1, kj, 32);
    const float* vr = vb + (size_t)(kj + 32) * D_;
    o0 += p * vr[lane]; o1 += p * vr[32 + lane];
    o2 += p * vr[64 + lane]; o3 += p * vr[96 + lane];
  }
  float* orow = o + ((size_t)(b * S_ + qi) * D_ + (size_t)h * HD_);
  orow[lane] = o0; orow[32 + lane] = o1; orow[64 + lane] = o2; orow[96 + lane] = o3;
}

// gate[i] = silu(gate[i]) * up[i]
__global__ void silu_mul_kernel(float* __restrict__ gate, const float* __restrict__ up, int n) {
  int i = blockIdx.x * blockDim.x + threadIdx.x;
  if (i >= n) return;
  float g = gate[i];
  gate[i] = (g / (1.f + expf(-g))) * up[i];
}

// logits[b, c] = hn_flat[b, :] . om_w[c, :] + om_b[c]
__global__ __launch_bounds__(256)
void om_kernel(const float* __restrict__ hn, const float* __restrict__ omw,
               const float* __restrict__ omb, float* __restrict__ logits) {
  __shared__ float red[8];
  int b = blockIdx.x / 5, c = blockIdx.x % 5;
  const float* hr = hn + (size_t)b * (S_ * D_);
  const float* wr = omw + (size_t)c * (S_ * D_);
  float ss = 0.f;
  for (int i = threadIdx.x; i < S_ * D_; i += 256) ss += hr[i] * wr[i];
  float tot = block_reduce_sum_256(ss, red);
  if (threadIdx.x == 0) logits[b * 5 + c] = tot + omb[c];
}

__global__ void softmax5_kernel(const float* __restrict__ logits, float* __restrict__ out) {
  int b = threadIdx.x;
  if (b >= 32) return;
  float v[5], m = -1e30f;
  #pragma unroll
  for (int c = 0; c < 5; ++c) { v[c] = logits[b * 5 + c]; m = fmaxf(m, v[c]); }
  float s = 0.f;
  #pragma unroll
  for (int c = 0; c < 5; ++c) { v[c] = expf(v[c] - m); s += v[c]; }
  #pragma unroll
  for (int c = 0; c < 5; ++c) out[b * 5 + c] = v[c] / s;
}

// ---------------------------------------------------------------------------
// Orchestration
// ---------------------------------------------------------------------------
extern "C" void kernel_launch(void* const* d_in, const int* in_sizes, int n_in,
                              void* d_out, int out_size, void* d_ws, size_t ws_size,
                              hipStream_t stream) {
  const float* x        = (const float*)d_in[0];
  const float* conv1_w  = (const float*)d_in[1];
  const float* conv1_b  = (const float*)d_in[2];
  const float* conv2_w  = (const float*)d_in[3];
  const float* conv2_b  = (const float*)d_in[4];
  const float* it_w     = (const float*)d_in[5];
  const float* it_b     = (const float*)d_in[6];
  const float* tokproj  = (const float*)d_in[7];
  const float* ln1_w    = (const float*)d_in[8];
  const float* wq       = (const float*)d_in[9];
  const float* wk       = (const float*)d_in[10];
  const float* wv       = (const float*)d_in[11];
  const float* wo       = (const float*)d_in[12];
  const float* ln2_w    = (const float*)d_in[13];
  const float* w_gate   = (const float*)d_in[14];
  const float* w_up     = (const float*)d_in[15];
  const float* w_down   = (const float*)d_in[16];
  const float* fnorm_w  = (const float*)d_in[17];
  const float* om_w     = (const float*)d_in[18];
  const float* om_b     = (const float*)d_in[19];
  float* out = (float*)d_out;

  // Workspace layout (fp32 elements)
  float* ws = (float*)d_ws;
  size_t off = 0;
  const size_t TOKD = (size_t)MT_ * D_;       // 8,388,608
  const size_t TOKF = (size_t)MT_ * FF_;      // 22,544,384
  float* hs   = ws + off; off += TOKD;
  float* hn   = ws + off; off += TOKD;
  float* qb   = ws + off; off += TOKD;
  float* kb   = ws + off; off += TOKD;
  float* vb   = ws + off; off += TOKD;
  float* ob   = ws + off; off += TOKD;
  float* gate = ws + off; off += TOKF;
  float* up   = ws + off; off += TOKF;
  float* h1   = ws + off; off += (size_t)32 * 32 * 31 * 31;
  float* h2   = ws + off; off += (size_t)32 * 64 * 14 * 14;
  float* tok  = ws + off; off += MT_;
  float* cost = ws + off; off += (size_t)S_ * HD_;
  float* sint = ws + off; off += (size_t)S_ * HD_;
  float* lgts = ws + off; off += 160;
  (void)ws_size; (void)in_sizes; (void)n_in; (void)out_size;

  // --- Conv embedding + input transformation ---
  conv1_pool<<<(32 * 32 * 31 * 31 + 255) / 256, 256, 0, stream>>>(x, conv1_w, conv1_b, h1);
  conv2_pool<<<(32 * 64 * 14 * 14 + 255) / 256, 256, 0, stream>>>(h1, conv2_w, conv2_b, h2);
  it_kernel<<<MT_, 256, 0, stream>>>(h2, it_w, it_b, tok);
  embed_kernel<<<(int)((TOKD + 255) / 256), 256, 0, stream>>>(tok, tokproj, hs);
  rope_tables<<<(S_ * 64 + 255) / 256, 256, 0, stream>>>(cost, sint);

  const dim3 gemm_blk(256);
  const dim3 grid_dd(D_ / 64, MT_ / 128);     // N=4096
  const dim3 grid_df(FF_ / 64, MT_ / 128);    // N=11008

  for (int l = 0; l < L_; ++l) {
    const float* wq_l = wq + (size_t)l * D_ * D_;
    const float* wk_l = wk + (size_t)l * D_ * D_;
    const float* wv_l = wv + (size_t)l * D_ * D_;
    const float* wo_l = wo + (size_t)l * D_ * D_;
    const float* wg_l = w_gate + (size_t)l * FF_ * D_;
    const float* wu_l = w_up   + (size_t)l * FF_ * D_;
    const float* wd_l = w_down + (size_t)l * D_ * FF_;

    rmsnorm_kernel<<<MT_, 256, 0, stream>>>(hs, ln1_w + (size_t)l * D_, hn);
    gemm_bf16_tn<<<grid_dd, gemm_blk, 0, stream>>>(hn, wq_l, nullptr, qb, MT_, D_, D_);
    gemm_bf16_tn<<<grid_dd, gemm_blk, 0, stream>>>(hn, wk_l, nullptr, kb, MT_, D_, D_);
    gemm_bf16_tn<<<grid_dd, gemm_blk, 0, stream>>>(hn, wv_l, nullptr, vb, MT_, D_, D_);
    rope_apply<<<(MT_ * H_ * 64 + 255) / 256, 256, 0, stream>>>(qb, kb, cost, sint);
    attn_kernel<<<B_ * H_ * S_, 32, 0, stream>>>(qb, kb, vb, ob);
    gemm_bf16_tn<<<grid_dd, gemm_blk, 0, stream>>>(ob, wo_l, hs, hs, MT_, D_, D_);   // + residual

    rmsnorm_kernel<<<MT_, 256, 0, stream>>>(hs, ln2_w + (size_t)l * D_, hn);
    gemm_bf16_tn<<<grid_df, gemm_blk, 0, stream>>>(hn, wg_l, nullptr, gate, MT_, FF_, D_);
    gemm_bf16_tn<<<grid_df, gemm_blk, 0, stream>>>(hn, wu_l, nullptr, up,   MT_, FF_, D_);
    silu_mul_kernel<<<(int)((TOKF + 255) / 256), 256, 0, stream>>>(gate, up, (int)TOKF);
    gemm_bf16_tn<<<grid_dd, gemm_blk, 0, stream>>>(gate, wd_l, hs, hs, MT_, D_, FF_); // + residual
  }

  rmsnorm_kernel<<<MT_, 256, 0, stream>>>(hs, fnorm_w, hn);
  om_kernel<<<32 * 5, 256, 0, stream>>>(hn, om_w, om_b, lgts);
  softmax5_kernel<<<1, 32, 0, stream>>>(lgts, out);
}
